// W8A8B8O32Linear_32968168964173
// MI455X (gfx1250) — compile-verified, matrix operand order checked by reading, not run on hardware
//
#include <hip/hip_runtime.h>
#include <stdint.h>

typedef __attribute__((ext_vector_type(8))) int          v8i;
typedef __attribute__((ext_vector_type(4))) int          v4i;
typedef __attribute__((ext_vector_type(4))) unsigned int v4u;

#define M_DIM 8192
#define N_DIM 4096
#define K_DIM 4096
#define BM 128
#define BN 128
#define BK 128
#define KT (K_DIM / BK)          // 32 K-steps
#define TILE_BYTES (BM * BK)     // 16 KB per LDS tile
#define KT64 (K_DIM / 64)        // 64 k-tiles of width 64
#define ROW_BYTES (KT64 * 1024)  // 64 KB per 16-row band in swizzled layout

// ---------------- byte packing via v_perm_b32 ----------------
__device__ __forceinline__ int pack4_perm(int a, int b, int c, int d) {
  unsigned lo = __builtin_amdgcn_perm((unsigned)b, (unsigned)a, 0x0C0C0400u);
  unsigned hi = __builtin_amdgcn_perm((unsigned)d, (unsigned)c, 0x04000C0Cu);
  return (int)(lo | hi);
}

// ------- pass 1: quantize + swizzle into WMMA-fragment order ---------------
// Output layout: [row16-band][ktile64][lane(32)][32 bytes], so each lane's
// 16x64 IU8 operand fragment (8 dwords) is contiguous in memory/LDS.
// Per the ISA A-layout, lane<16 covers K-dwords {0,1,4,5,8,9,12,13}, lane>=16
// the same +2 (element +8), row = lane&15.
__global__ __launch_bounds__(256)
void swizzle_x_kernel(const int* __restrict__ x, int* __restrict__ xs) {
  const int gid  = blockIdx.x * 256 + threadIdx.x;
  const int lane = gid & 31;
  const int tile = gid >> 5;              // tm * KT64 + tk
  const int tk   = tile & (KT64 - 1);
  const int tm   = tile >> 6;             // log2(KT64) = 6
  const int row  = tm * 16 + (lane & 15);
  const int* src = x + (size_t)row * K_DIM + tk * 64 + (lane >> 4) * 8;
  int o[8];
  #pragma unroll
  for (int j = 0; j < 4; ++j) {
    int4 p0 = *(const int4*)(src + j * 16);
    int4 p1 = *(const int4*)(src + j * 16 + 4);
    o[2 * j]     = pack4_perm(p0.x, p0.y, p0.z, p0.w);
    o[2 * j + 1] = pack4_perm(p1.x, p1.y, p1.z, p1.w);
  }
  int4* dst = (int4*)xs + (size_t)tile * 64 + lane * 2;
  dst[0] = make_int4(o[0], o[1], o[2], o[3]);
  dst[1] = make_int4(o[4], o[5], o[6], o[7]);
}

__global__ __launch_bounds__(256)
void swizzle_w_kernel(const float* __restrict__ w, int* __restrict__ wsw) {
  const int gid  = blockIdx.x * 256 + threadIdx.x;
  const int lane = gid & 31;
  const int tile = gid >> 5;
  const int tk   = tile & (KT64 - 1);
  const int tn   = tile >> 6;
  const int row  = tn * 16 + (lane & 15);
  const float* src = w + (size_t)row * K_DIM + tk * 64 + (lane >> 4) * 8;
  int o[8];
  #pragma unroll
  for (int j = 0; j < 4; ++j) {
    float4 p0 = *(const float4*)(src + j * 16);
    float4 p1 = *(const float4*)(src + j * 16 + 4);
    o[2 * j]     = pack4_perm(__float2int_rn(p0.x), __float2int_rn(p0.y),
                              __float2int_rn(p0.z), __float2int_rn(p0.w));
    o[2 * j + 1] = pack4_perm(__float2int_rn(p1.x), __float2int_rn(p1.y),
                              __float2int_rn(p1.z), __float2int_rn(p1.w));
  }
  int4* dst = (int4*)wsw + (size_t)tile * 64 + lane * 2;
  dst[0] = make_int4(o[0], o[1], o[2], o[3]);
  dst[1] = make_int4(o[4], o[5], o[6], o[7]);
}

// ---------------- TDM 2D tile load (D# per ISA 08_async_tensor §8.3-8.5) ----
__device__ __forceinline__ void tdm_load_2d(uint32_t lds_off, const void* gptr,
                                            uint32_t tensor_d0, uint32_t tensor_d1,
                                            uint32_t d0_stride,
                                            uint32_t tile_d0, uint32_t tile_d1) {
  const uint64_t ga = (uint64_t)(uintptr_t)gptr;
  v4u g0;
  g0[0] = 1u;                                 // count=1, user descriptor
  g0[1] = lds_off;                            // lds_addr (bytes)
  g0[2] = (uint32_t)ga;                       // global_addr[31:0]
  g0[3] = (uint32_t)(ga >> 32) | (2u << 30);  // global_addr[56:32] | type=2
  v8i g1;
  g1[0] = 0;                                  // wg_mask=0, data_size=1B
  g1[1] = (int)((tensor_d0 & 0xFFFFu) << 16);
  g1[2] = (int)((tensor_d0 >> 16) | ((tensor_d1 & 0xFFFFu) << 16));
  g1[3] = (int)((tensor_d1 >> 16) | (tile_d0 << 16));
  g1[4] = (int)(tile_d1 & 0xFFFFu);           // tile_dim2 = 0
  g1[5] = (int)d0_stride;
  g1[6] = 0;
  g1[7] = 0;
  v4i g2 = {0, 0, 0, 0};
  v4i g3 = {0, 0, 0, 0};
  v8i g4 = {0, 0, 0, 0, 0, 0, 0, 0};
  __builtin_amdgcn_tensor_load_to_lds(g0, g1, g2, g3, g4, 0);
}

// ---------------- fragment fetch: 2x ds_load_b128, static offsets ----------
__device__ __forceinline__ v8i load_frag(const int8_t* p) {
  int4 lo = *(const int4*)p;
  int4 hi = *(const int4*)(p + 16);
  v8i r;
  r[0] = lo.x; r[1] = lo.y; r[2] = lo.z; r[3] = lo.w;
  r[4] = hi.x; r[5] = hi.y; r[6] = hi.z; r[7] = hi.w;
  return r;
}

template <int S>
__device__ __forceinline__ void compute_stage(const int8_t* laneA,
                                              const int8_t* laneB,
                                              v8i (&acc)[2][4]) {
  // Load ALL fragments for both K-halves first (no WAR reuse between WMMAs).
  v8i af[2][2];  // [mt][h]
  v8i bf[4][2];  // [nt][h]
  #pragma unroll
  for (int mt = 0; mt < 2; ++mt)
    #pragma unroll
    for (int h = 0; h < 2; ++h)
      af[mt][h] = load_frag(laneA + S * TILE_BYTES + (mt * 2 + h) * 1024);
  #pragma unroll
  for (int nt = 0; nt < 4; ++nt)
    #pragma unroll
    for (int h = 0; h < 2; ++h)
      bf[nt][h] = load_frag(laneB + S * TILE_BYTES + (nt * 2 + h) * 1024);

  #pragma unroll
  for (int h = 0; h < 2; ++h)
    #pragma unroll
    for (int mt = 0; mt < 2; ++mt)
      #pragma unroll
      for (int nt = 0; nt < 4; ++nt)
        acc[mt][nt] = __builtin_amdgcn_wmma_i32_16x16x64_iu8(
            true, af[mt][h], true, bf[nt][h], acc[mt][nt], false, false);
}

// ---------------- pass 2: int8 WMMA GEMM with TDM double buffering ---------
__global__ __launch_bounds__(256)
void w8a8_wmma_gemm(const int8_t* __restrict__ xs,   // swizzled [M/16][K/64][1024]
                    const int8_t* __restrict__ wsw,  // swizzled [N/16][K/64][1024]
                    const float* __restrict__ bias,
                    const float* __restrict__ alpha_p,
                    float* __restrict__ out) {
  // [A s0 | A s1 | B s0 | B s1], 16 KB each; smem at LDS offset 0.
  __shared__ __align__(128) int8_t smem[4 * TILE_BYTES];

  const int tid  = threadIdx.x;
  const int lane = tid & 31;
  const int wave = tid >> 5;
  const int wm   = wave & 3;    // 4 M-groups of 32 rows (2 m16-tiles)
  const int wn   = wave >> 2;   // 2 N-groups of 64 cols (4 n16-tiles)

  const int mBlock = blockIdx.y * BM;
  const int nBlock = blockIdx.x * BN;

  const float alphaV = alpha_p[0];

  v8i acc[2][4];
  #pragma unroll
  for (int mt = 0; mt < 2; ++mt)
    #pragma unroll
    for (int nt = 0; nt < 4; ++nt)
      #pragma unroll
      for (int e = 0; e < 8; ++e)
        acc[mt][nt][e] = 0;

  const uint32_t smemBase = (uint32_t)(uintptr_t)&smem[0];
  // Per-wave LDS fragment bases; all further offsets are compile-time consts.
  const int8_t* laneA = smem + lane * 32 + wm * 4096;
  const int8_t* laneB = smem + 2 * TILE_BYTES + lane * 32 + wn * 8192;

  auto issue_tdm = [&](int kt, int s) {
    const int8_t* gA = xs + (size_t)(mBlock / 16) * ROW_BYTES + (size_t)kt * 2048;
    const int8_t* gB = wsw + (size_t)(nBlock / 16) * ROW_BYTES + (size_t)kt * 2048;
    // 8 rows x 2048 bytes, row stride 64 KB -> contiguous 16 KB in LDS.
    tdm_load_2d(smemBase + (uint32_t)s * TILE_BYTES, gA,
                ROW_BYTES, M_DIM / 16, ROW_BYTES, 2048, 8);
    tdm_load_2d(smemBase + (uint32_t)(2 + s) * TILE_BYTES, gB,
                ROW_BYTES, N_DIM / 16, ROW_BYTES, 2048, 8);
  };

  if (wave == 0) {
    issue_tdm(0, 0);
    __builtin_amdgcn_s_wait_tensorcnt(0);
  }
  __syncthreads();

  for (int kt = 0; kt < KT; kt += 2) {
    if (wave == 0 && kt + 1 < KT) issue_tdm(kt + 1, 1);
    compute_stage<0>(laneA, laneB, acc);
    if (wave == 0 && kt + 1 < KT) __builtin_amdgcn_s_wait_tensorcnt(0);
    __syncthreads();

    if (wave == 0 && kt + 2 < KT) issue_tdm(kt + 2, 0);
    compute_stage<1>(laneA, laneB, acc);
    if (wave == 0 && kt + 2 < KT) __builtin_amdgcn_s_wait_tensorcnt(0);
    __syncthreads();
  }

  // Epilogue: D layout — VGPR i holds (M = base + i [+8 hi lanes], N = lane&15)
  const int laneN = lane & 15;
  #pragma unroll
  for (int nt = 0; nt < 4; ++nt) {
    const int n = nBlock + wn * 64 + nt * 16 + laneN;
    const float bn = bias[n];
    #pragma unroll
    for (int mt = 0; mt < 2; ++mt) {
      const int mrow = mBlock + wm * 32 + mt * 16 + (lane >> 4) * 8;
      #pragma unroll
      for (int i = 0; i < 8; ++i) {
        float v = (float)acc[mt][nt][i] * alphaV + bn;
        v = rintf(v);
        v = fminf(127.0f, fmaxf(-128.0f, v));
        out[(size_t)(mrow + i) * N_DIM + n] = v;
      }
    }
  }
}

extern "C" void kernel_launch(void* const* d_in, const int* in_sizes, int n_in,
                              void* d_out, int out_size, void* d_ws, size_t ws_size,
                              hipStream_t stream) {
  (void)in_sizes; (void)n_in; (void)out_size; (void)ws_size;
  const int*   x     = (const int*)d_in[0];    // [4,2048,4096] int32
  const float* w     = (const float*)d_in[1];  // [4096,4096] fp32
  const float* bias  = (const float*)d_in[2];  // [1,4096] fp32
  const float* alpha = (const float*)d_in[3];  // scalar
  float*       out   = (float*)d_out;          // [8192,4096]

  int8_t* xs  = (int8_t*)d_ws;                         // 32 MB swizzled x
  int8_t* wsw = xs + (size_t)M_DIM * K_DIM;            // +16 MB swizzled w

  // Pass 1: quantize + swizzle. One thread per (16x64 tile, lane).
  swizzle_x_kernel<<<(M_DIM / 16) * KT64 * 32 / 256, 256, 0, stream>>>(x, (int*)xs);
  swizzle_w_kernel<<<(N_DIM / 16) * KT64 * 32 / 256, 256, 0, stream>>>(w, (int*)wsw);

  // Pass 2: WMMA GEMM with TDM staging.
  dim3 grid(N_DIM / BN, M_DIM / BM);  // (32, 64)
  w8a8_wmma_gemm<<<grid, 256, 0, stream>>>(xs, wsw, bias, alpha, out);
}
